// SimulatorGNN_49280454754549
// MI455X (gfx1250) — matrix operand order repaired and checked by reading, not compiled, not run on hardware
//
#include <hip/hip_runtime.h>
#include <hip/hip_bf16.h>

// ============================================================================
// SimulatorGNN for MI455X (gfx1250, wave32, WMMA).
//
// Roofline: ~1e12 FLOPs (edge MLPs dominate), ~6-8 GB of gather/scatter
// traffic at bf16 => balanced at bf16 WMMA with fp32 accumulation.
// All 128-wide GEMMs run on v_wmma_f32_16x16x32_bf16; LayerNorm and the
// segment_sum (fp32 atomicAdd scatter) stay in fp32.
// Gather staging uses GLOBAL_LOAD_ASYNC_TO_LDS_B128 (ASYNCcnt-tracked direct
// global->LDS copy); the builtin's param type is v4i in AS1/AS3 (learned from
// the round-3 diagnostic).
// ============================================================================

typedef __bf16 bf16_t;
typedef __attribute__((ext_vector_type(16))) __bf16 v16bf;
typedef __attribute__((ext_vector_type(8)))  float  v8f;

struct __align__(16) BF8 { bf16_t h[8]; };

#define LN_EPS 1e-5f

#if defined(__has_builtin)
#if __has_builtin(__builtin_amdgcn_global_load_async_to_lds_b128) && \
    __has_builtin(__builtin_amdgcn_s_wait_asynccnt)
#define USE_ASYNC_LDS 1
#endif
#endif
#ifndef USE_ASYNC_LDS
#define USE_ASYNC_LDS 0
#endif

#if USE_ASYNC_LDS
typedef int v4i_async __attribute__((vector_size(4 * sizeof(int))));
typedef __attribute__((address_space(1))) v4i_async* gasync_ptr_t;
typedef __attribute__((address_space(3))) v4i_async* lasync_ptr_t;
#endif

// ---------------------------------------------------------------------------
// One GEMM stage over a 32x128 output tile, 8 waves (one 16-col block each),
// 2 row-blocks per wave.  X: LDS bf16 [32][RS].  Wt: global bf16 [128][K]
// (transposed, row = output column).  Writes bias(+ReLU) result to LDS as
// bf16 (stride 128) or fp32 (stride 128).
//
// Register layouts per CDNA5 ISA 7.12.2 (wave32):
//   A 16x32 bf16 : lane<16 -> M=lane; VGPR v<4: K=2v,2v+1 (+8 if lane>=16),
//                  v>=4: K=16+2(v-4) (+8 if lane>=16)
//   B 32x16 bf16 : lane%16 -> N; K per VGPR v and lane half same as A
//   C 16x16 f32  : lane%16 -> N; VGPR r -> M = r + 8*(lane/16)
// ---------------------------------------------------------------------------
__device__ inline void gemm_tile(const bf16_t* __restrict__ X, int RS, int K,
                                 const bf16_t* __restrict__ Wt,
                                 const float* __restrict__ bias, int nbias,
                                 bool relu,
                                 bf16_t* __restrict__ outB,
                                 float* __restrict__ outF)
{
    const int tid  = threadIdx.x;
    const int lane = tid & 31;
    const int w    = tid >> 5;
    const int half = lane >> 4;
    const int mr   = lane & 15;
    const int col  = (w << 4) | mr;

    v8f acc0 = {};
    v8f acc1 = {};
    const bf16_t* wrow = Wt + (size_t)col * K;

    for (int kb = 0; kb < K; kb += 32) {
        v16bf a0, a1, bm;
#pragma unroll
        for (int v = 0; v < 8; ++v) {
            int kA = kb + ((v < 4) ? (2 * v + 8 * half)
                                   : (16 + 2 * (v - 4) + 8 * half));
            int kB = kA;
            a0[2 * v]     = X[mr * RS + kA];
            a0[2 * v + 1] = X[mr * RS + kA + 1];
            a1[2 * v]     = X[(mr + 16) * RS + kA];
            a1[2 * v + 1] = X[(mr + 16) * RS + kA + 1];
            bm[2 * v]     = wrow[kB];
            bm[2 * v + 1] = wrow[kB + 1];
        }
        acc0 = __builtin_amdgcn_wmma_f32_16x16x32_bf16(false, a0, false, bm,
                                                       (short)0, acc0, false, false);
        acc1 = __builtin_amdgcn_wmma_f32_16x16x32_bf16(false, a1, false, bm,
                                                       (short)0, acc1, false, false);
    }

    float bv = (bias && col < nbias) ? bias[col] : 0.0f;
#pragma unroll
    for (int r = 0; r < 8; ++r) {
        int m = r + (half << 3);
        float v0 = acc0[r] + bv;
        float v1 = acc1[r] + bv;
        if (relu) { v0 = fmaxf(v0, 0.0f); v1 = fmaxf(v1, 0.0f); }
        if (outB) {
            outB[m * 128 + col]        = (bf16_t)v0;
            outB[(m + 16) * 128 + col] = (bf16_t)v1;
        } else {
            outF[m * 128 + col]        = v0;
            outF[(m + 16) * 128 + col] = v1;
        }
    }
}

// ---------------------------------------------------------------------------
// Unified 3-layer MLP kernel over 32-row tiles.
//   SMALLK : layer0 is a thin fp32 VALU layer (K_in <= 6), else a wide bf16
//            WMMA GEMM over K = NSRC*128 with per-source gather indices.
//   OUTMODE: 0 = LayerNorm then store bf16 rows (16B vector stores)
//            1 = LayerNorm then fp32 atomicAdd scatter (segment_sum/residual)
//            2 = no LN, small head (nOut<=16), direct fp32 store (wave 0)
// ---------------------------------------------------------------------------
template <bool SMALLK, int NSRC, int OUTMODE>
__global__ void mlp3_kernel(int M,
                            const float* __restrict__ x, int kin,
                            const float* __restrict__ W0f,
                            const float* __restrict__ b0f,
                            const void* s0, const int* i0, int f0,
                            const void* s1, const int* i1, int f1,
                            const void* s2, const int* i2, int f2,
                            const bf16_t* __restrict__ Wt0, const float* __restrict__ b0,
                            const bf16_t* __restrict__ Wt1, const float* __restrict__ b1,
                            const bf16_t* __restrict__ Wt2, const float* __restrict__ b2,
                            int nb2,
                            const float* __restrict__ lng,
                            const float* __restrict__ lnb,
                            bf16_t* __restrict__ outB,
                            float* __restrict__ outF,
                            const int* __restrict__ outIdx,
                            int nOut)
{
    constexpr int K1 = NSRC * 128;
    __shared__ bf16_t xs[32 * K1];
    __shared__ bf16_t hs[32 * 128];
    __shared__ float  cs[32 * 128];
    __shared__ float  xrow[32 * 8];      // encoder layer-0 inputs (kin <= 6)
    __shared__ float  ps[256], ps2[256]; // LN partial sums
    __shared__ float  mu[32];
    __shared__ float  rsg[32];

    const int tid   = threadIdx.x;
    const int tile0 = blockIdx.x * 32;

    if (SMALLK) {
        // stage the thin input block once (32 x kin fp32)
        for (int id = tid; id < 32 * kin; id += 256) {
            int m = id / kin, k = id - m * kin;
            int rg = tile0 + m;
            xrow[m * 8 + k] = (rg < M) ? x[(size_t)rg * kin + k] : 0.0f;
        }
        __syncthreads();
        // layer0: thin fp32 layer (K_in in {1,4,6}) -> hs (relu, bf16)
        for (int id = tid; id < 32 * 128; id += 256) {
            int m = id >> 7, n = id & 127;
            int rg = tile0 + m;
            float a = 0.0f;
            if (rg < M) {
                a = b0f[n];
                for (int k = 0; k < kin; ++k)
                    a += xrow[m * 8 + k] * W0f[k * 128 + n];
                a = fmaxf(a, 0.0f);
            }
            hs[id] = (bf16_t)a;
        }
        __syncthreads();
    } else {
        // stage concatenated (gathered) inputs into xs: 16-byte chunks.
        // bf16 sources go via GLOBAL_LOAD_ASYNC_TO_LDS_B128 when available.
        const void* sp[3] = { s0, s1, s2 };
        const int*  ip[3] = { i0, i1, i2 };
        const int   fp[3] = { f0, f1, f2 };
        constexpr int CPR = K1 / 8;   // 16-byte chunks per row
        for (int id = tid; id < 32 * CPR; id += 256) {
            int m  = id / CPR;
            int ch = id - m * CPR;
            int sidx = ch >> 4;       // 16 chunks per 128-wide source
            int cc8  = (ch & 15) << 3;
            int rg = tile0 + m;
            bf16_t* ldst = xs + m * K1 + ch * 8;
            if (rg < M) {
                int row = ip[sidx] ? ip[sidx][rg] : rg;
                if (fp[sidx]) {
                    const float4* pf = (const float4*)
                        ((const float*)sp[sidx] + (size_t)row * 128 + cc8);
                    float4 f0v = pf[0];
                    float4 f1v = pf[1];
                    BF8 o;
                    o.h[0] = (bf16_t)f0v.x; o.h[1] = (bf16_t)f0v.y;
                    o.h[2] = (bf16_t)f0v.z; o.h[3] = (bf16_t)f0v.w;
                    o.h[4] = (bf16_t)f1v.x; o.h[5] = (bf16_t)f1v.y;
                    o.h[6] = (bf16_t)f1v.z; o.h[7] = (bf16_t)f1v.w;
                    *(BF8*)ldst = o;
                } else {
                    const bf16_t* gsrc = (const bf16_t*)sp[sidx] + (size_t)row * 128 + cc8;
#if USE_ASYNC_LDS
                    __builtin_amdgcn_global_load_async_to_lds_b128(
                        (gasync_ptr_t)(void*)gsrc, (lasync_ptr_t)(void*)ldst, 0, 0);
#else
                    *(BF8*)ldst = *(const BF8*)gsrc;
#endif
                }
            } else {
                BF8 o;
#pragma unroll
                for (int q = 0; q < 8; ++q) o.h[q] = (bf16_t)0.0f;
                *(BF8*)ldst = o;
            }
        }
#if USE_ASYNC_LDS
        __builtin_amdgcn_s_wait_asynccnt(0);
#endif
        __syncthreads();
        // layer0: wide GEMM (K = NSRC*128) -> hs (relu)
        gemm_tile(xs, K1, K1, Wt0, b0, 128, true, hs, nullptr);
        __syncthreads();
    }

    // layer1: 128 -> 128 (relu) -> xs (stride 128)
    gemm_tile(hs, 128, 128, Wt1, b1, 128, true, xs, nullptr);
    __syncthreads();

    if (OUTMODE == 2) {
        // layer2 small head: wave 0 computes the single 16-col block
        if (tid < 32) {
            const int lane = tid;
            const int half = lane >> 4;
            const int mr   = lane & 15;
            const int col  = mr;
            v8f acc0 = {};
            v8f acc1 = {};
            const bf16_t* wrow = Wt2 + (size_t)col * 128;
            for (int kb = 0; kb < 128; kb += 32) {
                v16bf a0, a1, bm;
#pragma unroll
                for (int v = 0; v < 8; ++v) {
                    int kA = kb + ((v < 4) ? (2 * v + 8 * half)
                                           : (16 + 2 * (v - 4) + 8 * half));
                    int kB = kA;
                    a0[2 * v]     = xs[mr * 128 + kA];
                    a0[2 * v + 1] = xs[mr * 128 + kA + 1];
                    a1[2 * v]     = xs[(mr + 16) * 128 + kA];
                    a1[2 * v + 1] = xs[(mr + 16) * 128 + kA + 1];
                    bm[2 * v]     = wrow[kB];
                    bm[2 * v + 1] = wrow[kB + 1];
                }
                acc0 = __builtin_amdgcn_wmma_f32_16x16x32_bf16(false, a0, false, bm,
                                                               (short)0, acc0, false, false);
                acc1 = __builtin_amdgcn_wmma_f32_16x16x32_bf16(false, a1, false, bm,
                                                               (short)0, acc1, false, false);
            }
            float bv = (b2 && col < nb2) ? b2[col] : 0.0f;
#pragma unroll
            for (int r = 0; r < 8; ++r) {
                int m = r + (half << 3);
                if (col < nOut) {
                    int rg0 = tile0 + m;
                    int rg1 = tile0 + 16 + m;
                    if (rg0 < M) outF[(size_t)rg0 * nOut + col] = acc0[r] + bv;
                    if (rg1 < M) outF[(size_t)rg1 * nOut + col] = acc1[r] + bv;
                }
            }
        }
    } else {
        // layer2: 128 -> 128 (no relu) -> cs fp32, then LayerNorm
        gemm_tile(xs, 128, 128, Wt2, b2, nb2, false, nullptr, cs);
        __syncthreads();
        // parallel LN reduction: 8 threads per row, 16 elems each
        {
            int m   = tid >> 3;
            int seg = tid & 7;
            float s = 0.0f, s2 = 0.0f;
#pragma unroll
            for (int q = 0; q < 16; ++q) {
                float v = cs[m * 128 + seg * 16 + q];
                s  += v;
                s2 += v * v;
            }
            ps[tid]  = s;
            ps2[tid] = s2;
        }
        __syncthreads();
        if ((tid & 7) == 0) {
            int m = tid >> 3;
            float S = 0.0f, S2 = 0.0f;
#pragma unroll
            for (int q = 0; q < 8; ++q) { S += ps[tid + q]; S2 += ps2[tid + q]; }
            float m_  = S * (1.0f / 128.0f);
            float var = S2 * (1.0f / 128.0f) - m_ * m_;
            mu[m]  = m_;
            rsg[m] = rsqrtf(fmaxf(var, 0.0f) + LN_EPS);
        }
        __syncthreads();
        if (OUTMODE == 0) {
            // vectorized bf16 stores (16B)
            for (int id = tid; id < 32 * 16; id += 256) {
                int m  = id >> 4;
                int c8 = (id & 15) << 3;
                int rg = tile0 + m;
                if (rg >= M) continue;
                BF8 o;
#pragma unroll
                for (int q = 0; q < 8; ++q) {
                    int c = c8 + q;
                    float v = (cs[m * 128 + c] - mu[m]) * rsg[m] * lng[c] + lnb[c];
                    o.h[q] = (bf16_t)v;
                }
                *(BF8*)(outB + (size_t)rg * 128 + c8) = o;
            }
        } else {
            // fp32 atomic scatter (segment_sum / residual accumulation)
            for (int id = tid; id < 32 * 128; id += 256) {
                int m = id >> 7, c = id & 127;
                int rg = tile0 + m;
                if (rg >= M) continue;
                float v = (cs[id] - mu[m]) * rsg[m] * lng[c] + lnb[c];
                int row = outIdx ? outIdx[rg] : rg;
                atomicAdd(&outF[(size_t)row * 128 + c], v);
            }
        }
    }
}

// ------------------------- utility kernels ---------------------------------
__global__ void k_zero_f32(float* __restrict__ p, long long n)
{
    long long i = (long long)blockIdx.x * blockDim.x + threadIdx.x;
    if (i < n) p[i] = 0.0f;
}

__global__ void k_f32_from_bf16(float* __restrict__ d, const bf16_t* __restrict__ s,
                                long long n)
{
    long long i = (long long)blockIdx.x * blockDim.x + threadIdx.x;
    if (i < n) d[i] = (float)s[i];
}

__global__ void k_bf16_from_f32(bf16_t* __restrict__ d, const float* __restrict__ s,
                                long long n)
{
    long long i = (long long)blockIdx.x * blockDim.x + threadIdx.x;
    if (i < n) d[i] = (bf16_t)s[i];
}

// fp32 [K][N] -> bf16 transposed [Npad][K] (rows = output columns, zero-padded)
__global__ void k_convT(bf16_t* __restrict__ dst, const float* __restrict__ src,
                        int K, int N, int Npad)
{
    int i = blockIdx.x * blockDim.x + threadIdx.x;
    if (i >= Npad * K) return;
    int n = i / K;
    int k = i - n * K;
    dst[i] = (n < N) ? (bf16_t)src[(size_t)k * N + n] : (bf16_t)0.0f;
}

// ============================================================================
// Host orchestration
// ============================================================================
extern "C" void kernel_launch(void* const* d_in, const int* in_sizes, int n_in,
                              void* d_out, int out_size, void* d_ws, size_t ws_size,
                              hipStream_t stream)
{
    (void)in_sizes; (void)n_in; (void)out_size; (void)ws_size;

    const int N_W = 4000, N_O = 40000, N_F = 4000;
    const int E_W2O = 200000, E_W2F = 40000, E_OO = 640000, E_FO = 200000;

    const float* x_world  = (const float*)d_in[0];
    const float* x_object = (const float*)d_in[1];
    const float* x_floor  = (const float*)d_in[2];
    const float* ea_w2o   = (const float*)d_in[3];
    const float* ea_w2f   = (const float*)d_in[4];
    const float* ea_oo    = (const float*)d_in[5];
    const float* ea_fo    = (const float*)d_in[6];
    const int*   ei_w2o   = (const int*)d_in[7];   // [0:E)=src, [E:2E)=dst
    const int*   ei_w2f   = (const int*)d_in[8];
    const int*   ei_oo    = (const int*)d_in[9];
    const int*   ei_fo    = (const int*)d_in[10];

    // Param leaf bases (jax pytree flatten: sorted dict keys; lists in order).
    // MLP w/ LN  leaves: b0,b1,b2,ln_b,ln_g,w0,w1,w2  (8)
    // MLP no LN  leaves: b0,b1,b2,w0,w1,w2            (6)
    const int LB_DEC_CONTACT = 11;
    const int LB_DEC_OBJECT  = 17;
    const int LB_ENC_CONTACT = 23;
    const int LB_ENC_FLOOR   = 31;
    const int LB_ENC_OBJECT  = 39;
    const int LB_ENC_W2F     = 47;
    const int LB_ENC_W2O     = 55;
    const int LB_ENC_WORLD   = 63;
    const int LB_PROC        = 71;   // 4 steps x [fo,oo,w2f,w2o] x [edge,node] x 8

    // ---- workspace bump allocator (deterministic) ----
    size_t off = 0;
    auto alloc = [&](size_t bytes) -> void* {
        void* p = (char*)d_ws + off;
        off += (bytes + 255) & ~(size_t)255;
        return p;
    };

    bf16_t* h_w   = (bf16_t*)alloc((size_t)N_W   * 128 * sizeof(bf16_t));
    bf16_t* h_o   = (bf16_t*)alloc((size_t)N_O   * 128 * sizeof(bf16_t));
    bf16_t* h_f   = (bf16_t*)alloc((size_t)N_F   * 128 * sizeof(bf16_t));
    bf16_t* e_w2o = (bf16_t*)alloc((size_t)E_W2O * 128 * sizeof(bf16_t));
    bf16_t* e_w2f = (bf16_t*)alloc((size_t)E_W2F * 128 * sizeof(bf16_t));
    bf16_t* e_oo  = (bf16_t*)alloc((size_t)E_OO  * 128 * sizeof(bf16_t));
    bf16_t* e_fo  = (bf16_t*)alloc((size_t)E_FO  * 128 * sizeof(bf16_t));
    float*  agg   = (float*) alloc((size_t)N_O   * 128 * sizeof(float));
    float*  accO  = (float*) alloc((size_t)N_O   * 128 * sizeof(float));
    float*  accF  = (float*) alloc((size_t)N_F   * 128 * sizeof(float));

    auto leaf = [&](int i) { return (const float*)d_in[i]; };

    auto cvt = [&](int li, int K, int N, int Npad) -> const bf16_t* {
        bf16_t* dst = (bf16_t*)alloc((size_t)Npad * K * sizeof(bf16_t));
        int total = Npad * K;
        k_convT<<<dim3((total + 255) / 256), dim3(256), 0, stream>>>(
            dst, leaf(li), K, N, Npad);
        return dst;
    };

    struct Mlp {
        const float*  W0f; const float* b0;
        const bf16_t* W0t;
        const bf16_t* W1t; const float* b1;
        const bf16_t* W2t; const float* b2;
        const float*  g;   const float* bb;
    };

    auto mkEnc = [&](int base) {
        Mlp m{};
        m.W0f = leaf(base + 5); m.b0 = leaf(base + 0);
        m.W1t = cvt(base + 6, 128, 128, 128); m.b1 = leaf(base + 1);
        m.W2t = cvt(base + 7, 128, 128, 128); m.b2 = leaf(base + 2);
        m.g = leaf(base + 4); m.bb = leaf(base + 3);
        return m;
    };
    auto mkProc = [&](int base, int K0) {
        Mlp m{};
        m.W0t = cvt(base + 5, K0, 128, 128);  m.b0 = leaf(base + 0);
        m.W1t = cvt(base + 6, 128, 128, 128); m.b1 = leaf(base + 1);
        m.W2t = cvt(base + 7, 128, 128, 128); m.b2 = leaf(base + 2);
        m.g = leaf(base + 4); m.bb = leaf(base + 3);
        return m;
    };
    auto mkDec = [&](int base, int K0, int nOut) {
        Mlp m{};
        m.W0t = cvt(base + 3, K0, 128, 128);  m.b0 = leaf(base + 0);
        m.W1t = cvt(base + 4, 128, 128, 128); m.b1 = leaf(base + 1);
        m.W2t = cvt(base + 5, 128, nOut, 16); m.b2 = leaf(base + 2);
        return m;
    };

    auto zero = [&](float* p, long long n) {
        k_zero_f32<<<dim3((unsigned)((n + 255) / 256)), dim3(256), 0, stream>>>(p, n);
    };
    auto toF32 = [&](float* d, const bf16_t* s, long long n) {
        k_f32_from_bf16<<<dim3((unsigned)((n + 255) / 256)), dim3(256), 0, stream>>>(d, s, n);
    };
    auto toB16 = [&](bf16_t* d, const float* s, long long n) {
        k_bf16_from_f32<<<dim3((unsigned)((n + 255) / 256)), dim3(256), 0, stream>>>(d, s, n);
    };

    // ---------------- encoders ----------------
    auto encLaunch = [&](const float* x, int M, int kin, const Mlp& p, bf16_t* out) {
        mlp3_kernel<true, 1, 0><<<dim3((M + 31) / 32), dim3(256), 0, stream>>>(
            M, x, kin, p.W0f, p.b0,
            nullptr, nullptr, 0, nullptr, nullptr, 0, nullptr, nullptr, 0,
            nullptr, nullptr,
            p.W1t, p.b1, p.W2t, p.b2, 128,
            p.g, p.bb,
            out, nullptr, nullptr, 128);
    };

    Mlp encWorld   = mkEnc(LB_ENC_WORLD);
    Mlp encObject  = mkEnc(LB_ENC_OBJECT);
    Mlp encFloor   = mkEnc(LB_ENC_FLOOR);
    Mlp encW2O     = mkEnc(LB_ENC_W2O);
    Mlp encW2F     = mkEnc(LB_ENC_W2F);
    Mlp encContact = mkEnc(LB_ENC_CONTACT);

    encLaunch(x_world,  N_W,   4, encWorld,   h_w);
    encLaunch(x_object, N_O,   6, encObject,  h_o);
    encLaunch(x_floor,  N_F,   1, encFloor,   h_f);
    encLaunch(ea_w2o,   E_W2O, 4, encW2O,     e_w2o);
    encLaunch(ea_w2f,   E_W2F, 1, encW2F,     e_w2f);
    encLaunch(ea_oo,    E_OO,  4, encContact, e_oo);
    encLaunch(ea_fo,    E_FO,  4, encContact, e_fo);

    // ---------------- processor: 4 message-passing steps ----------------
    auto edgeLaunch = [&](int E, const bf16_t* hdst, const int* dstIdx,
                          const bf16_t* hsrc, const int* srcIdx,
                          const bf16_t* efeat, const Mlp& p, float* aggOut) {
        mlp3_kernel<false, 3, 1><<<dim3((E + 31) / 32), dim3(256), 0, stream>>>(
            E, nullptr, 0, nullptr, nullptr,
            (const void*)hdst, dstIdx, 0,
            (const void*)hsrc, srcIdx, 0,
            (const void*)efeat, nullptr, 0,
            p.W0t, p.b0, p.W1t, p.b1, p.W2t, p.b2, 128,
            p.g, p.bb,
            nullptr, aggOut, dstIdx, 128);
    };
    auto nodeLaunch = [&](int nDst, const bf16_t* hdst, const float* aggIn,
                          const Mlp& p, float* accOut) {
        mlp3_kernel<false, 2, 1><<<dim3((nDst + 31) / 32), dim3(256), 0, stream>>>(
            nDst, nullptr, 0, nullptr, nullptr,
            (const void*)hdst, nullptr, 0,
            (const void*)aggIn, nullptr, 1,
            nullptr, nullptr, 0,
            p.W0t, p.b0, p.W1t, p.b1, p.W2t, p.b2, 128,
            p.g, p.bb,
            nullptr, accOut, nullptr, 128);
    };

    for (int s = 0; s < 4; ++s) {
        int sb = LB_PROC + s * 64;  // rel leaves: fo(+0), oo(+16), w2f(+32), w2o(+48)

        toF32(accO, h_o, (long long)N_O * 128);   // residual base
        toF32(accF, h_f, (long long)N_F * 128);

        struct Rel {
            int base;
            const bf16_t* hsrc; const bf16_t* hdst;
            int E; const int* ei; const bf16_t* ef;
            int nDst; float* acc;
        } rels[4] = {
            { sb + 0,  h_f, h_o, E_FO,  ei_fo,  e_fo,  N_O, accO },  // fo
            { sb + 16, h_o, h_o, E_OO,  ei_oo,  e_oo,  N_O, accO },  // oo
            { sb + 32, h_w, h_f, E_W2F, ei_w2f, e_w2f, N_F, accF },  // w2f
            { sb + 48, h_w, h_o, E_W2O, ei_w2o, e_w2o, N_O, accO },  // w2o
        };

        for (int r = 0; r < 4; ++r) {
            const Rel& R = rels[r];
            Mlp pe = mkProc(R.base + 0, 384);
            Mlp pn = mkProc(R.base + 8, 256);
            const int* srcIdx = R.ei;
            const int* dstIdx = R.ei + R.E;
            zero(agg, (long long)R.nDst * 128);
            edgeLaunch(R.E, R.hdst, dstIdx, R.hsrc, srcIdx, R.ef, pe, agg);
            nodeLaunch(R.nDst, R.hdst, agg, pn, R.acc);
        }

        toB16(h_o, accO, (long long)N_O * 128);
        toB16(h_f, accF, (long long)N_F * 128);
    }

    // ---------------- decoders ----------------
    float* outF       = (float*)d_out;
    float* out_obj    = outF;                       // N_O x 3
    float* out_con_oo = outF + (size_t)N_O * 3;     // E_OO x 1
    float* out_con_fo = out_con_oo + (size_t)E_OO;  // E_FO x 1

    Mlp decObj = mkDec(LB_DEC_OBJECT, 128, 3);
    Mlp decCon = mkDec(LB_DEC_CONTACT, 384, 1);

    mlp3_kernel<false, 1, 2><<<dim3((N_O + 31) / 32), dim3(256), 0, stream>>>(
        N_O, nullptr, 0, nullptr, nullptr,
        (const void*)h_o, nullptr, 0,
        nullptr, nullptr, 0, nullptr, nullptr, 0,
        decObj.W0t, decObj.b0, decObj.W1t, decObj.b1, decObj.W2t, decObj.b2, 3,
        nullptr, nullptr,
        nullptr, out_obj, nullptr, 3);

    // contact(oo): cat[h_o[src], h_o[dst], e_oo]
    mlp3_kernel<false, 3, 2><<<dim3((E_OO + 31) / 32), dim3(256), 0, stream>>>(
        E_OO, nullptr, 0, nullptr, nullptr,
        (const void*)h_o, ei_oo, 0,
        (const void*)h_o, ei_oo + E_OO, 0,
        (const void*)e_oo, nullptr, 0,
        decCon.W0t, decCon.b0, decCon.W1t, decCon.b1, decCon.W2t, decCon.b2, 1,
        nullptr, nullptr,
        nullptr, out_con_oo, nullptr, 1);

    // contact(fo): cat[h_f[src], h_o[dst], e_fo]
    mlp3_kernel<false, 3, 2><<<dim3((E_FO + 31) / 32), dim3(256), 0, stream>>>(
        E_FO, nullptr, 0, nullptr, nullptr,
        (const void*)h_f, ei_fo, 0,
        (const void*)h_o, ei_fo + E_FO, 0,
        (const void*)e_fo, nullptr, 0,
        decCon.W0t, decCon.b0, decCon.W1t, decCon.b1, decCon.W2t, decCon.b2, 1,
        nullptr, nullptr,
        nullptr, out_con_fo, nullptr, 1);
}
